// DeepCLIP_75514114998500
// MI455X (gfx1250) — matrix operand order repaired and checked by reading, not compiled, not run on hardware
//
#include <hip/hip_runtime.h>
#include <hip/hip_bf16.h>
#include <cmath>

#define SEQ  4096
#define XLEN 16384
#define NB   256

// ---- CDNA5 WMMA types (16x16x32 bf16 -> f32) ----
typedef __bf16 bf16_t;
typedef bf16_t v16bf __attribute__((ext_vector_type(16)));
typedef float  v8f   __attribute__((ext_vector_type(8)));
typedef unsigned int v4u __attribute__((ext_vector_type(4)));

union AFrag { v16bf v; v4u q[2]; unsigned short u[16]; };

__device__ __forceinline__ unsigned short f2bf(float f) {
  unsigned int u = __float_as_uint(f);
  unsigned int r = u + 0x7FFFu + ((u >> 16) & 1u);   // round-to-nearest-even
  return (unsigned short)(r >> 16);
}
__device__ __forceinline__ unsigned int pk2bf(float lo, float hi) {
  return (unsigned int)f2bf(lo) | ((unsigned int)f2bf(hi) << 16);
}

// gfx1250 hardware tanh (TRANS pipe), branch-free.
__device__ __forceinline__ float tanh_fast(float x) {
#if __has_builtin(__builtin_amdgcn_tanhf)
  return __builtin_amdgcn_tanhf(x);
#else
  float r;
  asm("v_tanh_f32 %0, %1\n\tv_nop" : "=v"(r) : "v"(x));  // v_nop: TRANS RAW slot
  return r;
#endif
}
__device__ __forceinline__ float sig_fast(float x) {
  return fmaf(0.5f, tanh_fast(0.5f * x), 0.5f);
}

// Wave-internal LDS producer->consumer ordering: emits s_wait_dscnt, no barrier.
// All staging buffers are private to one wave, so no cross-wave barrier needed.
__device__ __forceinline__ void lds_fence() {
  __builtin_amdgcn_fence(__ATOMIC_ACQ_REL, "workgroup");
}

// ---------------------------------------------------------------------------
// Kernel 1: per-(batch,filter) max of conv output; zero the batch accumulators.
// ---------------------------------------------------------------------------
__global__ void __launch_bounds__(256)
k1_maxconv(const float* __restrict__ x,
           const float* __restrict__ w4, const float* __restrict__ w5,
           const float* __restrict__ w6, const float* __restrict__ w7,
           const float* __restrict__ w8,
           float* __restrict__ wsmax, float* __restrict__ wsacc) {
  const int b = blockIdx.x, tid = threadIdx.x;
  const float* wp[5] = {w4, w5, w6, w7, w8};
  float cw[5][8];
  for (int f = 0; f < 5; ++f)
    for (int k = 0; k < 8; ++k) cw[f][k] = (k < 4 + f) ? wp[f][k] : 0.0f;

  float mf[5] = {-1e30f, -1e30f, -1e30f, -1e30f, -1e30f};
  const float* xb = x + (size_t)b * XLEN;
  for (int i = tid; i < SEQ; i += 256) {
    float xv[8];
#pragma unroll
    for (int k = 0; k < 8; ++k) { int idx = 4 * i + k; xv[k] = (idx < XLEN) ? xb[idx] : 0.0f; }
#pragma unroll
    for (int f = 0; f < 5; ++f) {
      int Lf = (f == 0) ? SEQ : (SEQ - 1);
      if (i < Lf) {
        float s = 0.0f;
#pragma unroll
        for (int k = 0; k < 8; ++k) s += cw[f][k] * xv[k];   // cw zero-padded
        mf[f] = fmaxf(mf[f], s);
      }
    }
  }
  __shared__ float red[256];
  for (int f = 0; f < 5; ++f) {
    red[tid] = mf[f];
    __syncthreads();
    for (int off = 128; off > 0; off >>= 1) {
      if (tid < off) red[tid] = fmaxf(red[tid], red[tid + off]);
      __syncthreads();
    }
    if (tid == 0) wsmax[b * 5 + f] = red[0];
    __syncthreads();
  }
  if (tid == 0) wsacc[b] = 0.0f;
}

// ---------------------------------------------------------------------------
// Kernel 2: fused feature recompute + bidirectional LSTM + profile reduction.
// Block = one 16-sample batch tile; wave0 = forward scan, wave1 = reverse scan.
// The two waves are fully decoupled (disjoint LDS slices, no barriers).
// ---------------------------------------------------------------------------
__global__ void __launch_bounds__(64)
k2_lstm(const float* __restrict__ x,
        const float* __restrict__ w4, const float* __restrict__ w5,
        const float* __restrict__ w6, const float* __restrict__ w7,
        const float* __restrict__ w8,
        const float* __restrict__ Wih_f, const float* __restrict__ Whh_f,
        const float* __restrict__ bih_f, const float* __restrict__ bhh_f,
        const float* __restrict__ Wih_r, const float* __restrict__ Whh_r,
        const float* __restrict__ bih_r, const float* __restrict__ bhh_r,
        const float* __restrict__ wsmax, float* __restrict__ wsacc) {
  // Per-wave LDS staging (index [w]): features / h-state as bf16 [m][K],
  // gates as f32 [m][gate].
  __shared__ __align__(16) unsigned short featw[2][16][32];
  __shared__ __align__(16) unsigned short hw[2][16][32];
  __shared__ __align__(16) float gws[2][16][48];

  const int T    = blockIdx.x;
  const int w    = threadIdx.x >> 5;       // 0 = forward, 1 = reverse
  const int lane = threadIdx.x & 31;
  const bool rev = (w == 1);
  const int M     = lane & 15;             // sample row this lane serves in A frag
  const int khalf = lane >> 4;             // 0: K 0-7,16-23; 1: K 8-15,24-31 (B frag)
  const int Kbase = khalf * 16;
  const int b1    = khalf * 8;             // A-frag first K group: 0 or 8
  const int c16   = M;                     // gate column within N-tile (C/D frag)

  const float* Wih = rev ? Wih_r : Wih_f;
  const float* Whh = rev ? Whh_r : Whh_f;
  const float* bih = rev ? bih_r : bih_f;
  const float* bhh = rev ? bhh_r : bhh_f;

  // Loop-invariant B fragments (K x N bf16, ISA B striping) + bias C fragments.
  AFrag bfi[3], bfh[3];
  v8f biasC[3];
#pragma unroll
  for (int n = 0; n < 3; ++n) {
    int g = n * 16 + c16;
#pragma unroll
    for (int j = 0; j < 16; ++j) {
      int K = (j < 8) ? (b1 + j) : (16 + b1 + (j - 8));
      bfi[n].u[j] = f2bf((g < 40 && K < 24) ? Wih[g * 24 + K] : 0.0f);
      bfh[n].u[j] = f2bf((g < 40 && K < 10) ? Whh[g * 10 + K] : 0.0f);
    }
    float bv = (g < 40) ? (bih[g] + bhh[g]) : 0.0f;
#pragma unroll
    for (int r = 0; r < 8; ++r) biasC[n][r] = bv;
  }

  // Per-sample constants (lanes 0..15 own sample M).
  const int bidx = T * 16 + M;
  const float* xb = x + (size_t)bidx * XLEN;
  float cw[5][8];
  {
    const float* wp[5] = {w4, w5, w6, w7, w8};
    for (int f = 0; f < 5; ++f)
      for (int k = 0; k < 8; ++k) cw[f][k] = (k < 4 + f) ? wp[f][k] : 0.0f;
  }
  float mfv[5];
  for (int f = 0; f < 5; ++f) mfv[f] = wsmax[bidx * 5 + f];

  // Zero this wave's bf16 staging slices (feat K-pad region, h state = 0).
  for (int idx = lane; idx < 16 * 32; idx += 32) {
    ((unsigned short*)featw[w])[idx] = 0;
    ((unsigned short*)hw[w])[idx]    = 0;
  }
  lds_fence();

  // Sliding 12-float x window: W12[j] = x[4s-4+j] for current s (lanes 0..15).
  const int s0 = rev ? (SEQ - 1) : 0;
  float W12[12];
#pragma unroll
  for (int j = 0; j < 12; ++j) {
    int idx = 4 * s0 - 4 + j;
    W12[j] = (idx >= 0 && idx < XLEN && lane < 16) ? xb[idx] : 0.0f;
  }

  float creg[5] = {0, 0, 0, 0, 0};         // cell state: 5 (m,hh) pairs per lane
  float accB = 0.0f;                        // sum_s hsum(s)*xs4(s), even lanes
  float xs4  = 0.0f;
  const int hh0 = (lane & 1) * 5;           // gate-combine mapping
  const int mC  = lane >> 1;

  for (int t = 0; t < SEQ; ++t) {
    const int s = rev ? (SEQ - 1 - t) : t;

    // (a) Features for position s from the register window (lanes 0..15).
    if (lane < 16) {
      xs4 = W12[4] + W12[5] + W12[6] + W12[7];
      unsigned int fpk[12];
      float fv[24];
#pragma unroll
      for (int f = 0; f < 5; ++f) {
        const int fs = 4 + f;
        float Sc = 0.0f, Sp = 0.0f;
#pragma unroll
        for (int k = 0; k < 8; ++k) { Sc += cw[f][k] * W12[4 + k]; Sp += cw[f][k] * W12[k]; }
        // scatter algebra: only conv taps i=s and i=s-1 reach position s
        float vc = (f == 0 || s < SEQ - 1) ? (Sc + mfv[f]) * (Sc + mfv[f]) : 0.0f;
        float vp = (s >= 1) ? (Sp + mfv[f]) * (Sp + mfv[f]) : 0.0f;
#pragma unroll
        for (int v = 0; v < 4; ++v) fv[4 * f + v] = vc + ((4 + v) < fs ? vp : 0.0f);
      }
#pragma unroll
      for (int v = 0; v < 4; ++v) fv[20 + v] = W12[4 + v];
#pragma unroll
      for (int p = 0; p < 12; ++p) fpk[p] = pk2bf(fv[2 * p], fv[2 * p + 1]);
      unsigned int* frow = (unsigned int*)&featw[w][M][0];
#pragma unroll
      for (int p = 0; p < 12; ++p) frow[p] = fpk[p];
      // prefetch a cacheline ahead in scan direction
      __builtin_prefetch(xb + 4 * s + (rev ? -24 : 24), 0, 3);
    }
    lds_fence();   // featw (this iter) + hw (prev iter) visible wave-wide

    // (c,d) Gather A fragments (ISA 16-bit A layout) and run 6 WMMAs.
    AFrag ax, ah;
    ax.q[0] = *(const v4u*)&featw[w][M][b1];
    ax.q[1] = *(const v4u*)&featw[w][M][b1 + 16];
    ah.q[0] = *(const v4u*)&hw[w][M][b1];
    ah.q[1] = *(const v4u*)&hw[w][M][b1 + 16];

    v8f acc[3];
#pragma unroll
    for (int n = 0; n < 3; ++n) {
      v8f c0 = biasC[n];
      c0 = __builtin_amdgcn_wmma_f32_16x16x32_bf16(false, ax.v, false, bfi[n].v,
                                                   (short)0, c0, false, false);
      c0 = __builtin_amdgcn_wmma_f32_16x16x32_bf16(false, ah.v, false, bfh[n].v,
                                                   (short)0, c0, false, false);
      acc[n] = c0;
    }

    // (e) Scatter D fragments (lane holds col c16, rows r+8*khalf) to [m][gate].
#pragma unroll
    for (int n = 0; n < 3; ++n) {
      const int g = n * 16 + c16;
#pragma unroll
      for (int r = 0; r < 8; ++r) gws[w][r + 8 * khalf][g] = acc[n][r];
    }
    lds_fence();   // gates visible wave-wide

    // (g) LSTM nonlinearity + state update: lane owns (m=lane/2, hh0..hh0+4).
    float hsum = 0.0f;
#pragma unroll
    for (int q = 0; q < 5; ++q) {
      const int hh = hh0 + q;
      float ig = gws[w][mC][hh];
      float fg = gws[w][mC][10 + hh];
      float gg = gws[w][mC][20 + hh];
      float og = gws[w][mC][30 + hh];
      creg[q] = sig_fast(fg) * creg[q] + sig_fast(ig) * tanh_fast(gg);
      float h = sig_fast(og) * tanh_fast(creg[q]);
      hsum += h;
      hw[w][mC][hh] = f2bf(h);              // next step's A_h fragment source
    }

    // (i) pairwise hsum combine + profile accumulation via cross-lane shuffles.
    float htot = hsum + __shfl_xor(hsum, 1);
    float xs4m = __shfl(xs4, mC);            // xs4 lives in lane mC (0..15)
    if ((lane & 1) == 0) accB += htot * xs4m;

    // (k) slide the x window for the next position (lanes 0..15).
    if (lane < 16) {
      if (!rev) {
#pragma unroll
        for (int j = 0; j < 8; ++j) W12[j] = W12[j + 4];
        const int nb = 4 * s + 8;
        if (nb + 3 < XLEN) {
          float4 nv = *(const float4*)(xb + nb);
          W12[8] = nv.x; W12[9] = nv.y; W12[10] = nv.z; W12[11] = nv.w;
        } else {
          W12[8] = W12[9] = W12[10] = W12[11] = 0.0f;
        }
      } else {
#pragma unroll
        for (int j = 11; j >= 4; --j) W12[j] = W12[j - 4];
        const int nb = 4 * s - 8;
        if (nb >= 0) {
          float4 nv = *(const float4*)(xb + nb);
          W12[0] = nv.x; W12[1] = nv.y; W12[2] = nv.z; W12[3] = nv.w;
        } else {
          W12[0] = W12[1] = W12[2] = W12[3] = 0.0f;
        }
      }
    }
  }

  // fwd + rev both contribute; even lanes own sample mC.
  if ((lane & 1) == 0) atomicAdd(&wsacc[T * 16 + mC], accB);
}

// ---------------------------------------------------------------------------
// Kernel 3: final sigmoid.
// ---------------------------------------------------------------------------
__global__ void k3_out(const float* __restrict__ wsacc, float* __restrict__ out) {
  const int b = threadIdx.x;
  out[b] = 1.0f / (1.0f + expf(-wsacc[b]));
}

extern "C" void kernel_launch(void* const* d_in, const int* in_sizes, int n_in,
                              void* d_out, int out_size, void* d_ws, size_t ws_size,
                              hipStream_t stream) {
  (void)in_sizes; (void)n_in; (void)out_size; (void)ws_size;
  const float* x     = (const float*)d_in[0];
  const float* w4    = (const float*)d_in[1];
  const float* w5    = (const float*)d_in[2];
  const float* w6    = (const float*)d_in[3];
  const float* w7    = (const float*)d_in[4];
  const float* w8    = (const float*)d_in[5];
  const float* Wih_f = (const float*)d_in[6];
  const float* Whh_f = (const float*)d_in[7];
  const float* bih_f = (const float*)d_in[8];
  const float* bhh_f = (const float*)d_in[9];
  const float* Wih_r = (const float*)d_in[10];
  const float* Whh_r = (const float*)d_in[11];
  const float* bih_r = (const float*)d_in[12];
  const float* bhh_r = (const float*)d_in[13];

  float* wsacc = (float*)d_ws;        // 256 f32
  float* wsmax = wsacc + NB;          // 256*5 f32  (total scratch ~6 KB)

  k1_maxconv<<<NB, 256, 0, stream>>>(x, w4, w5, w6, w7, w8, wsmax, wsacc);
  k2_lstm<<<16, 64, 0, stream>>>(x, w4, w5, w6, w7, w8,
                                 Wih_f, Whh_f, bih_f, bhh_f,
                                 Wih_r, Whh_r, bih_r, bhh_r,
                                 wsmax, wsacc);
  k3_out<<<1, NB, 0, stream>>>(wsacc, (float*)d_out);
}
